// ContrastiveLoss_89498528514555
// MI455X (gfx1250) — matrix operand order repaired and checked by reading, not compiled, not run on hardware
//
#include <hip/hip_runtime.h>

#define N_DATA   1200000
#define FEAT     128
#define BATCH    512
#define N_NEG    4096
#define INV_TEMP (1.0f / 0.07f)

typedef __attribute__((ext_vector_type(16))) _Float16 v16h;
typedef __attribute__((ext_vector_type(8)))  float    v8f;

__device__ __forceinline__ float wave_sum(float v) {
#pragma unroll
  for (int o = 16; o > 0; o >>= 1) v += __shfl_xor(v, o, 32);
  return v;
}

// ---------------- Kernel 1: L2-normalize s,t rows; pos_sim -----------------
__global__ __launch_bounds__(256) void nk_ctl(const float* __restrict__ s,
                                              const float* __restrict__ t,
                                              float* __restrict__ s_n,
                                              float* __restrict__ t_n,
                                              float* __restrict__ pos) {
  int row  = blockIdx.x * (blockDim.x >> 5) + (threadIdx.x >> 5);
  int lane = threadIdx.x & 31;
  if (row >= BATCH) return;
  const float* sp = s + row * FEAT;
  const float* tp = t + row * FEAT;
  float sv[4], tv[4], ss = 0.f, tt = 0.f;
#pragma unroll
  for (int i = 0; i < 4; ++i) {
    sv[i] = sp[lane + 32 * i]; tv[i] = tp[lane + 32 * i];
    ss += sv[i] * sv[i];       tt += tv[i] * tv[i];
  }
  ss = wave_sum(ss); tt = wave_sum(tt);
  float si = 1.f / fmaxf(sqrtf(ss), 1e-12f);
  float ti = 1.f / fmaxf(sqrtf(tt), 1e-12f);
  float dot = 0.f;
#pragma unroll
  for (int i = 0; i < 4; ++i) {
    float a = sv[i] * si, b = tv[i] * ti;
    s_n[row * FEAT + lane + 32 * i] = a;
    t_n[row * FEAT + lane + 32 * i] = b;
    dot += a * b;
  }
  dot = wave_sum(dot);
  if (lane == 0) pos[row] = dot * INV_TEMP;
}

// ------- Kernel 2: gather + WMMA dot products + online logsumexp -----------
__global__ __launch_bounds__(256) void simk_ctl(const float* __restrict__ neg,
                                                const int* __restrict__ nidx_all,
                                                const float* __restrict__ s_n,
                                                const float* __restrict__ pos,
                                                float* __restrict__ bl) {
  const int b    = blockIdx.x;
  const int tid  = threadIdx.x;
  const int wave = tid >> 5;
  const int lane = tid & 31;
  const int hi   = (lane >> 4) & 1;   // half-wave id
  const int m    = lane & 15;         // negative-row within tile / column id

  // B operand: all 16 columns replicate s[b] chunk.
  // 16-bit B layout (matches A): low lanes K 0-7 & 16-23, high lanes 8-15 & 24-31.
  const float* srow = s_n + b * FEAT;
  v16h Bv[4];
#pragma unroll
  for (int c = 0; c < 4; ++c) {
#pragma unroll
    for (int j = 0; j < 16; ++j)
      Bv[c][j] = (_Float16)srow[32 * c + hi * 8 + j + (j >= 8 ? 8 : 0)];
  }

  float mx = -1e30f, sm = 0.f;
  const int*  nidx    = nidx_all + (size_t)b * N_NEG;
  const bool  contrib = (m == 0);     // lanes 0 (rows 0-7) and 16 (rows 8-15)

  for (int tile = wave; tile < N_NEG / 16; tile += 8) {
    int idx = nidx[tile * 16 + m];
    const float* rowp = neg + (size_t)idx * FEAT;

    // prefetch next tile's row for this lane (global_prefetch_b8)
    int tn = tile + 8;
    if (tn < N_NEG / 16) {
      int idn = nidx[tn * 16 + m];
      __builtin_prefetch(neg + (size_t)idn * FEAT + hi * 8);
    }

    v8f acc = {0.f, 0.f, 0.f, 0.f, 0.f, 0.f, 0.f, 0.f};
#pragma unroll
    for (int c = 0; c < 4; ++c) {
      // A layout (16-bit, 16x32): low lanes K 0-7 & 16-23, high lanes 8-15 & 24-31
      const float4* p = (const float4*)(rowp + 32 * c + hi * 8);
      float4 a0 = p[0], a1 = p[1], a2 = p[4], a3 = p[5];
      v16h Av;
      Av[0]  = (_Float16)a0.x; Av[1]  = (_Float16)a0.y;
      Av[2]  = (_Float16)a0.z; Av[3]  = (_Float16)a0.w;
      Av[4]  = (_Float16)a1.x; Av[5]  = (_Float16)a1.y;
      Av[6]  = (_Float16)a1.z; Av[7]  = (_Float16)a1.w;
      Av[8]  = (_Float16)a2.x; Av[9]  = (_Float16)a2.y;
      Av[10] = (_Float16)a2.z; Av[11] = (_Float16)a2.w;
      Av[12] = (_Float16)a3.x; Av[13] = (_Float16)a3.y;
      Av[14] = (_Float16)a3.z; Av[15] = (_Float16)a3.w;
      acc = __builtin_amdgcn_wmma_f32_16x16x32_f16(false, Av, false, Bv[c],
                                                   (short)0, acc, false, false);
    }
    if (contrib) {
#pragma unroll
      for (int i = 0; i < 8; ++i) {
        float x = acc[i] * INV_TEMP;
        if (x <= mx) sm += __expf(x - mx);
        else { sm = sm * __expf(mx - x) + 1.f; mx = x; }
      }
    }
  }

  // combine (max, sumexp) across lanes
#pragma unroll
  for (int o = 16; o > 0; o >>= 1) {
    float mo = __shfl_xor(mx, o, 32);
    float so = __shfl_xor(sm, o, 32);
    float M  = fmaxf(mx, mo);
    sm = sm * __expf(mx - M) + so * __expf(mo - M);
    mx = M;
  }
  __shared__ float smx[8], ssm[8];
  if (lane == 0) { smx[wave] = mx; ssm[wave] = sm; }
  __syncthreads();
  if (tid == 0) {
    float M = smx[0], L = ssm[0];
    for (int w = 1; w < 8; ++w) {
      float Mn = fmaxf(M, smx[w]);
      L = L * __expf(M - Mn) + ssm[w] * __expf(smx[w] - Mn);
      M = Mn;
    }
    float p  = pos[b];
    float Mn = fmaxf(M, p);
    L = L * __expf(M - Mn) + __expf(p - Mn);
    M = Mn;
    bl[b] = (M + __logf(L)) - p;   // logsumexp - logits[:,0]
  }
}

// ------------- Kernel 3: deterministic fixed-order loss mean ---------------
__global__ __launch_bounds__(512) void lossred_ctl(const float* __restrict__ bl,
                                                   float* __restrict__ out) {
  __shared__ float sh[BATCH];
  int t = threadIdx.x;
  sh[t] = bl[t];
  __syncthreads();
  for (int s = BATCH / 2; s > 0; s >>= 1) {
    if (t < s) sh[t] += sh[t + s];
    __syncthreads();
  }
  if (t == 0) out[0] = sh[0] * (1.0f / BATCH);
}

// ------------- Kernel 4: stream-copy bank -> output (non-temporal) ---------
__global__ __launch_bounds__(256) void copyk_ctl(const float* __restrict__ src,
                                                 float* __restrict__ dst,
                                                 size_t n) {
  size_t stride = (size_t)gridDim.x * blockDim.x;
  size_t i = (size_t)blockIdx.x * blockDim.x + threadIdx.x;
  for (; i + 3 * stride < n; i += 4 * stride) {
    float a0 = __builtin_nontemporal_load(src + i);
    float a1 = __builtin_nontemporal_load(src + i + stride);
    float a2 = __builtin_nontemporal_load(src + i + 2 * stride);
    float a3 = __builtin_nontemporal_load(src + i + 3 * stride);
    __builtin_nontemporal_store(a0, dst + i);
    __builtin_nontemporal_store(a1, dst + i + stride);
    __builtin_nontemporal_store(a2, dst + i + 2 * stride);
    __builtin_nontemporal_store(a3, dst + i + 3 * stride);
  }
  for (; i < n; i += stride)
    __builtin_nontemporal_store(__builtin_nontemporal_load(src + i), dst + i);
}

// -------- Kernel 5: EMA scatter + renormalize (last-occurrence wins) -------
__global__ __launch_bounds__(128) void scatk_ctl(const float* __restrict__ neg,
                                                 const float* __restrict__ t_n,
                                                 const int* __restrict__ ind,
                                                 float* __restrict__ dst) {
  __shared__ int   skip;
  __shared__ float part[4];
  int b = blockIdx.x, tid = threadIdx.x;
  int myidx = ind[b];
  if (tid == 0) skip = 0;
  __syncthreads();
  int found = 0;
  for (int j = b + 1 + tid; j < BATCH; j += 128) found |= (ind[j] == myidx);
  if (found) skip = 1;   // benign race: all writers store 1
  __syncthreads();
  if (skip) return;      // a later batch row owns this bank slot
  float v  = 0.5f * neg[(size_t)myidx * FEAT + tid] + 0.5f * t_n[b * FEAT + tid];
  float ss = wave_sum(v * v);
  if ((tid & 31) == 0) part[tid >> 5] = ss;
  __syncthreads();
  float tot = part[0] + part[1] + part[2] + part[3];
  float inv = 1.f / fmaxf(sqrtf(tot), 1e-12f);
  dst[(size_t)myidx * FEAT + tid] = v * inv;
}

extern "C" void kernel_launch(void* const* d_in, const int* in_sizes, int n_in,
                              void* d_out, int out_size, void* d_ws, size_t ws_size,
                              hipStream_t stream) {
  const float* s   = (const float*)d_in[0];   // student_feat [512,128]
  const float* t   = (const float*)d_in[1];   // teacher_feat [512,128]
  const float* neg = (const float*)d_in[2];   // negatives [1.2M,128]
  const int*   idx = (const int*)d_in[3];     // indices [512]
  const int*   nix = (const int*)d_in[4];     // neg_indices [512,4096]
  float* out = (float*)d_out;                 // [0]=loss, [1..]=new_negatives

  float* ws  = (float*)d_ws;
  float* s_n = ws;                            // 512*128
  float* t_n = s_n + BATCH * FEAT;            // 512*128
  float* pos = t_n + BATCH * FEAT;            // 512
  float* bl  = pos + BATCH;                   // 512

  nk_ctl<<<BATCH / 8, 256, 0, stream>>>(s, t, s_n, t_n, pos);
  simk_ctl<<<BATCH, 256, 0, stream>>>(neg, nix, s_n, pos, bl);
  lossred_ctl<<<1, BATCH, 0, stream>>>(bl, out);
  size_t n = (size_t)N_DATA * FEAT;
  copyk_ctl<<<8192, 256, 0, stream>>>(neg, out + 1, n);
  scatk_ctl<<<BATCH, 128, 0, stream>>>(neg, t_n, idx, out + 1);
}
